// CDGTN_11957188952236
// MI455X (gfx1250) — matrix operand order repaired and checked by reading, not compiled, not run on hardware
//
#include <hip/hip_runtime.h>
#include <hip/hip_bf16.h>

typedef __attribute__((ext_vector_type(16))) _Float16 v16h;
typedef __attribute__((ext_vector_type(8)))  float    v8f;

#define NNODES 8192
#define DIM    768
#define CH     384
#define NEDGE  131072
#define NB     64
#define NL     128

// ---------------- elementwise converts ----------------
__global__ void k_cvt_f32_f16(const float* __restrict__ src, _Float16* __restrict__ dst, int n) {
    int i = blockIdx.x * 256 + threadIdx.x;
    if (i < n) dst[i] = (_Float16)src[i];
}

// W is [K x Nn] row-major f32; produce Wt [Nn x K] f16 (K contiguous per output row)
__global__ void k_trans_f16(const float* __restrict__ W, _Float16* __restrict__ Wt, int K, int Nn) {
    int idx = blockIdx.x * 256 + threadIdx.x;
    if (idx >= K * Nn) return;
    int k = idx / Nn, n = idx % Nn;
    Wt[(size_t)n * K + k] = (_Float16)W[idx];
}

__global__ void k_zero_i32(int* __restrict__ p, int n) {
    int i = blockIdx.x * 256 + threadIdx.x;
    if (i < n) p[i] = 0;
}

// ---------------- CSR build ----------------
__global__ void k_count(const int* __restrict__ dst, int* __restrict__ counts, int E) {
    int e = blockIdx.x * 256 + threadIdx.x;
    if (e < E) atomicAdd(&counts[dst[e]], 1);
}

__global__ void k_scan(const int* __restrict__ counts, int* __restrict__ offs) {
    __shared__ int part[1024];
    int t = threadIdx.x;
    int base = t * 8;
    int loc[8]; int s = 0;
#pragma unroll
    for (int j = 0; j < 8; ++j) { loc[j] = s; s += counts[base + j]; }
    part[t] = s;
    __syncthreads();
    for (int off = 1; off < 1024; off <<= 1) {
        int v = (t >= off) ? part[t - off] : 0;
        __syncthreads();
        part[t] += v;
        __syncthreads();
    }
    int prev = (t > 0) ? part[t - 1] : 0;
#pragma unroll
    for (int j = 0; j < 8; ++j) offs[base + j] = prev + loc[j];
    if (t == 1023) offs[NNODES] = part[1023];
}

__global__ void k_scatter(const int* __restrict__ src, const int* __restrict__ dst,
                          const int* __restrict__ offs, int* __restrict__ cur,
                          int* __restrict__ csrc, int E) {
    int e = blockIdx.x * 256 + threadIdx.x;
    if (e >= E) return;
    int d = dst[e];
    int p = offs[d] + atomicAdd(&cur[d], 1);
    csrc[p] = src[e];
}

// ---------------- WMMA GEMM ----------------
// C[M,Nn] = A[M,K](f16 row-major) x Wt[Nn,K](f16, K contiguous per out-col) + bias
// One wave computes a 32x64 macro-tile: 2 (M) x 4 (N) WMMA tiles, 8 wmma per K-step.
__global__ void k_gemm(const _Float16* __restrict__ A, const _Float16* __restrict__ Bt,
                       const float* __restrict__ bias, float* __restrict__ Cout,
                       int M, int Nn, int K) {
    int lane = threadIdx.x & 31;
    int wave = threadIdx.x >> 5;
    int tile = blockIdx.x * 8 + wave;
    int tilesN = Nn >> 6;                     // 64-wide macro columns
    int totTiles = (M >> 5) * tilesN;         // 32-tall macro rows
    if (tile >= totTiles) return;             // wave-uniform; active waves keep EXEC all-1
    int tm = tile / tilesN, tn = tile % tilesN;
    int m0 = tm << 5, n0 = tn << 6;
    int half = lane >> 4, r = lane & 15;

    v8f c[2][4];
#pragma unroll
    for (int j = 0; j < 4; ++j) {
        float bv = bias ? bias[n0 + 16 * j + r] : 0.f;
        v8f init = {bv, bv, bv, bv, bv, bv, bv, bv};
        c[0][j] = init;
        c[1][j] = init;
    }

    // A lane base: row m0(+16)+r, +8 halfs for lanes 16-31 (dword aligned: K even)
    const unsigned int* ap0 = (const unsigned int*)(A + (size_t)(m0 + r) * K + half * 8);
    const unsigned int* ap1 = (const unsigned int*)(A + (size_t)(m0 + 16 + r) * K + half * 8);
    // B lane bases: out-col n0+16j+r, +8 halfs for lanes 16-31
    const unsigned int* bp[4];
#pragma unroll
    for (int j = 0; j < 4; ++j)
        bp[j] = (const unsigned int*)(Bt + (size_t)(n0 + 16 * j + r) * K + half * 8);

    for (int k0 = 0; k0 < K; k0 += 32) {
        int db = k0 >> 1;
        // A 16x32 f16 layout: lanes 0-15 hold K{0..7,16..23}, lanes 16-31 K{8..15,24..31}
        union { v16h v; unsigned int u[8]; } a0, a1;
        a0.u[0] = ap0[db + 0];  a0.u[1] = ap0[db + 1];
        a0.u[2] = ap0[db + 2];  a0.u[3] = ap0[db + 3];
        a0.u[4] = ap0[db + 8];  a0.u[5] = ap0[db + 9];
        a0.u[6] = ap0[db + 10]; a0.u[7] = ap0[db + 11];
        a1.u[0] = ap1[db + 0];  a1.u[1] = ap1[db + 1];
        a1.u[2] = ap1[db + 2];  a1.u[3] = ap1[db + 3];
        a1.u[4] = ap1[db + 8];  a1.u[5] = ap1[db + 9];
        a1.u[6] = ap1[db + 10]; a1.u[7] = ap1[db + 11];
#pragma unroll
        for (int j = 0; j < 4; ++j) {
            // B 32x16 f16 layout: lanes 0-15 K{0..7,16..23}, lanes 16-31 K{8..15,24..31} (same as A)
            union { v16h v; unsigned int u[8]; } bf;
            const unsigned int* bj = bp[j];
            bf.u[0] = bj[db + 0]; bf.u[1] = bj[db + 1];
            bf.u[2] = bj[db + 2]; bf.u[3] = bj[db + 3];
            bf.u[4] = bj[db + 8]; bf.u[5] = bj[db + 9];
            bf.u[6] = bj[db + 10]; bf.u[7] = bj[db + 11];
            c[0][j] = __builtin_amdgcn_wmma_f32_16x16x32_f16(false, a0.v, false, bf.v,
                                                             (short)0, c[0][j], false, false);
            c[1][j] = __builtin_amdgcn_wmma_f32_16x16x32_f16(false, a1.v, false, bf.v,
                                                             (short)0, c[1][j], false, false);
        }
    }
    // C/D: VGPR i -> row base + half*8 + i, col n0 + 16j + r
#pragma unroll
    for (int i = 0; i < 2; ++i) {
        float* crow = Cout + (size_t)(m0 + 16 * i + half * 8) * Nn + n0 + r;
#pragma unroll
        for (int j = 0; j < 4; ++j)
#pragma unroll
            for (int v = 0; v < 8; ++v)
                crow[(size_t)v * Nn + 16 * j] = c[i][j][v];
    }
}

// ---------------- per-dst edge attention (online softmax over chunks of 64 edges) ----------------
__global__ void k_edge_attn(const float* __restrict__ Qb, const float* __restrict__ Kb,
                            const float* __restrict__ Vb, const float* __restrict__ Sb,
                            const int* __restrict__ offs, const int* __restrict__ csrc,
                            float* __restrict__ Hf32, _Float16* __restrict__ Hf16) {
    __shared__ float qld[DIM];
    __shared__ float lg[2][64];
    __shared__ float ww[2][64];
    __shared__ int   es[64];
    __shared__ float sM[2], sS[2], sScale[2];
    int n = blockIdx.x;
    int tid = threadIdx.x;
    for (int cc = tid; cc < DIM; cc += 256) qld[cc] = Qb[(size_t)n * DIM + cc];
    if (tid < 2) { sM[tid] = -3.0e38f; sS[tid] = 0.f; }
    __syncthreads();
    int start = offs[n], deg = offs[n + 1] - start;
    float acc0 = 0.f, acc1 = 0.f, acc2 = 0.f;
    const float rs = 0.05103103630798287f; // 1/sqrt(384)
    int lane = tid & 31, wv = tid >> 5;
    for (int chunk = 0; chunk < deg; chunk += 64) {
        int cnt = min(64, deg - chunk);
        // phase 1: per-edge logits (one wave per edge, lane-parallel dot)
        for (int e = wv; e < cnt; e += 8) {
            int sn = csrc[start + chunk + e];
            const float* kr = Kb + (size_t)sn * DIM;
            float d0 = 0.f, d1 = 0.f;
#pragma unroll
            for (int j = 0; j < 12; ++j) {
                int cc = lane + 32 * j;
                d0 += kr[cc] * qld[cc];
                d1 += kr[cc + CH] * qld[cc + CH];
            }
#pragma unroll
            for (int m = 16; m > 0; m >>= 1) {
                d0 += __shfl_xor(d0, m, 32);
                d1 += __shfl_xor(d1, m, 32);
            }
            if (lane == 0) { lg[0][e] = d0 * rs; lg[1][e] = d1 * rs; es[e] = sn; }
        }
        __syncthreads();
        // phase 2: online softmax state update (cheap serial, cnt <= 64)
        if (tid == 0) {
            for (int h = 0; h < 2; ++h) {
                float mo = sM[h], mn = mo;
                for (int e = 0; e < cnt; ++e) mn = fmaxf(mn, lg[h][e]);
                float sc = __expf(mo - mn);
                float s = sS[h] * sc;
                for (int e = 0; e < cnt; ++e) { float w = __expf(lg[h][e] - mn); ww[h][e] = w; s += w; }
                sM[h] = mn; sS[h] = s; sScale[h] = sc;
            }
        }
        __syncthreads();
        // phase 3: channel-parallel weighted gather of v[src]
        {
            int c0 = tid, c1 = tid + 256, c2 = tid + 512;
            int h0 = (c0 >= CH) ? 1 : 0, h1 = (c1 >= CH) ? 1 : 0, h2 = (c2 >= CH) ? 1 : 0;
            acc0 *= sScale[h0]; acc1 *= sScale[h1]; acc2 *= sScale[h2];
            for (int e = 0; e < cnt; ++e) {
                const float* vr = Vb + (size_t)es[e] * DIM;
                acc0 += ww[h0][e] * vr[c0];
                acc1 += ww[h1][e] * vr[c1];
                acc2 += ww[h2][e] * vr[c2];
            }
        }
        __syncthreads();
    }
    // finalize: alpha-normalize, +S residual, ReLU, write f32 (optional) + f16
    float accs[3] = {acc0, acc1, acc2};
#pragma unroll
    for (int i = 0; i < 3; ++i) {
        int cc = tid + 256 * i;
        int h = (cc >= CH) ? 1 : 0;
        float a = (deg > 0) ? accs[i] / sS[h] : 0.f;
        float o = a + Sb[(size_t)n * DIM + cc];
        o = fmaxf(o, 0.f);
        if (Hf32) Hf32[(size_t)n * DIM + cc] = o;
        Hf16[(size_t)n * DIM + cc] = (_Float16)o;
    }
}

// ---------------- pooling pieces ----------------
__global__ void k_gather_q(const _Float16* __restrict__ h16, _Float16* __restrict__ qh) {
    int idx = blockIdx.x * 256 + threadIdx.x;
    if (idx >= NB * DIM) return;
    int b = idx / DIM, c = idx % DIM;
    qh[idx] = h16[(size_t)b * NL * DIM + c];
}

__global__ void k_score(const float* __restrict__ pp, const float* __restrict__ qp,
                        const float* __restrict__ aw, const float* __restrict__ ab,
                        float* __restrict__ scores) {
    int node = blockIdx.x * 8 + (threadIdx.x >> 5);
    int lane = threadIdx.x & 31;
    int b = node >> 7;
    const float* pr = pp + (size_t)node * DIM;
    const float* qr = qp + (size_t)b * DIM;
    float s = 0.f;
#pragma unroll
    for (int j = 0; j < 24; ++j) {
        int c = lane + 32 * j;
        float x = pr[c] + qr[c];       // bias already in pp (GEMM bias)
        x = fmaxf(x, 0.f);             // relu
        s += x * aw[c];
    }
#pragma unroll
    for (int m = 16; m > 0; m >>= 1) s += __shfl_xor(s, m, 32);
    if (lane == 0) scores[node] = s + ab[0];
}

__global__ void k_pool(const float* __restrict__ scores, const float* __restrict__ Hp,
                       float* __restrict__ pooled) {
    __shared__ float sc[NL];
    __shared__ float sMax, sSum;
    int b = blockIdx.x, tid = threadIdx.x;
    if (tid < NL) sc[tid] = scores[b * NL + tid];
    __syncthreads();
    if (tid == 0) { float m = sc[0]; for (int l = 1; l < NL; ++l) m = fmaxf(m, sc[l]); sMax = m; }
    __syncthreads();
    if (tid < NL) sc[tid] = __expf(sc[tid] - sMax);
    __syncthreads();
    if (tid == 0) { float s = 0.f; for (int l = 0; l < NL; ++l) s += sc[l]; sSum = s; }
    __syncthreads();
    float inv = 1.f / sSum;
#pragma unroll
    for (int i = 0; i < 3; ++i) {
        int c = tid + 256 * i;
        const float* hb = Hp + (size_t)b * NL * DIM + c;
        float a = 0.f;
        for (int l = 0; l < NL; ++l) a += sc[l] * hb[(size_t)l * DIM];
        pooled[(size_t)b * DIM + c] = a * inv;
    }
}

__global__ void k_head(const float* __restrict__ pooled, const float* __restrict__ w1,
                       const float* __restrict__ b1, const float* __restrict__ w2,
                       const float* __restrict__ b2, float* __restrict__ out) {
    __shared__ float pin[DIM];
    __shared__ float t[DIM];
    __shared__ float red[256];
    __shared__ float o3[3];
    int b = blockIdx.x, tid = threadIdx.x;
    for (int c = tid; c < DIM; c += 256) pin[c] = pooled[(size_t)b * DIM + c];
    __syncthreads();
    for (int o = tid; o < DIM; o += 256) {
        float s = b1[o];
        for (int k = 0; k < DIM; ++k) s += pin[k] * w1[(size_t)k * DIM + o];
        t[o] = tanhf(s);
    }
    __syncthreads();
    for (int j = 0; j < 3; ++j) {
        float s = 0.f;
        for (int k = tid; k < DIM; k += 256) s += t[k] * w2[k * 3 + j];
        red[tid] = s;
        __syncthreads();
        for (int off = 128; off > 0; off >>= 1) {
            if (tid < off) red[tid] += red[tid + off];
            __syncthreads();
        }
        if (tid == 0) o3[j] = red[0] + b2[j];
        __syncthreads();
    }
    if (tid == 0) {
        float m = fmaxf(o3[0], fmaxf(o3[1], o3[2]));
        float s = __expf(o3[0] - m) + __expf(o3[1] - m) + __expf(o3[2] - m);
        float lse = m + __logf(s);
        out[b * 3 + 0] = o3[0] - lse;
        out[b * 3 + 1] = o3[1] - lse;
        out[b * 3 + 2] = o3[2] - lse;
    }
}

// ---------------- host launcher ----------------
extern "C" void kernel_launch(void* const* d_in, const int* in_sizes, int n_in,
                              void* d_out, int out_size, void* d_ws, size_t ws_size,
                              hipStream_t stream) {
    const int NS = NNODES * DIM;           // 6291456
    const int WE = DIM * DIM;              // 589824

    const float* x      = (const float*)d_in[0];
    const int*   eidx   = (const int*)d_in[1];
    const int*   esrc   = eidx;
    const int*   edst   = eidx + NEDGE;
    const float* wl[8]; const float* bl[8];
    for (int i = 0; i < 8; ++i) { wl[i] = (const float*)d_in[2 + 2 * i]; bl[i] = (const float*)d_in[3 + 2 * i]; }
    const float* atti_w = (const float*)d_in[18];
    const float* atti_b = (const float*)d_in[19];
    const float* atts_w = (const float*)d_in[20];
    const float* atts_b = (const float*)d_in[21];
    const float* fc1_w  = (const float*)d_in[22];
    const float* fc1_b  = (const float*)d_in[23];
    const float* fc2_w  = (const float*)d_in[24];
    const float* fc2_b  = (const float*)d_in[25];

    char* ws = (char*)d_ws;
    size_t off = 0;
    auto alloc = [&](size_t bytes) -> char* {
        char* p = ws + off;
        off += (bytes + 255) & ~(size_t)255;
        return p;
    };
    float*     bQ   = (float*)alloc((size_t)NS * 4);
    float*     bK   = (float*)alloc((size_t)NS * 4);
    float*     bV   = (float*)alloc((size_t)NS * 4);
    float*     bS   = (float*)alloc((size_t)NS * 4);
    float*     bH2  = (float*)alloc((size_t)NS * 4);
    _Float16*  bXH  = (_Float16*)alloc((size_t)NS * 2);
    _Float16*  bW[10];
    for (int i = 0; i < 10; ++i) bW[i] = (_Float16*)alloc((size_t)WE * 2);
    _Float16*  bQh    = (_Float16*)alloc((size_t)NB * DIM * 2);
    float*     bQpart = (float*)alloc((size_t)NB * DIM * 4);
    float*     bScore = (float*)alloc((size_t)NNODES * 4);
    float*     bPool  = (float*)alloc((size_t)NB * DIM * 4);
    int*       bCnt   = (int*)alloc((size_t)NNODES * 4);
    int*       bCur   = (int*)alloc((size_t)NNODES * 4);
    int*       bOffs  = (int*)alloc((size_t)(NNODES + 1) * 4);
    int*       bCsrc  = (int*)alloc((size_t)NEDGE * 4);
    (void)ws_size; (void)in_sizes; (void)n_in; (void)out_size;

    // f16 conversions / weight transposes
    k_cvt_f32_f16<<<(NS + 255) / 256, 256, 0, stream>>>(x, bXH, NS);
    for (int i = 0; i < 8; ++i)
        k_trans_f16<<<(WE + 255) / 256, 256, 0, stream>>>(wl[i], bW[i], DIM, DIM);
    k_trans_f16<<<(WE + 255) / 256, 256, 0, stream>>>(atti_w, bW[8], DIM, DIM);           // query half
    k_trans_f16<<<(WE + 255) / 256, 256, 0, stream>>>(atti_w + WE, bW[9], DIM, DIM);      // node half

    // CSR build
    k_zero_i32<<<(NNODES + 255) / 256, 256, 0, stream>>>(bCnt, NNODES);
    k_zero_i32<<<(NNODES + 255) / 256, 256, 0, stream>>>(bCur, NNODES);
    k_count<<<NEDGE / 256, 256, 0, stream>>>(edst, bCnt, NEDGE);
    k_scan<<<1, 1024, 0, stream>>>(bCnt, bOffs);
    k_scatter<<<NEDGE / 256, 256, 0, stream>>>(esrc, edst, bOffs, bCur, bCsrc, NEDGE);

    // 32x64 macro-tiles, 8 waves (macro-tiles) per block
    const int gemmBlocks  = ((NNODES / 32) * (DIM / 64) + 7) / 8;   // 384
    const int gemmBlocksQ = ((NB / 32) * (DIM / 64) + 7) / 8;       // 3

    // layer 1
    k_gemm<<<gemmBlocks, 256, 0, stream>>>(bXH, bW[0], bl[0], bQ, NNODES, DIM, DIM);
    k_gemm<<<gemmBlocks, 256, 0, stream>>>(bXH, bW[1], bl[1], bK, NNODES, DIM, DIM);
    k_gemm<<<gemmBlocks, 256, 0, stream>>>(bXH, bW[2], bl[2], bV, NNODES, DIM, DIM);
    k_gemm<<<gemmBlocks, 256, 0, stream>>>(bXH, bW[3], bl[3], bS, NNODES, DIM, DIM);
    k_edge_attn<<<NNODES, 256, 0, stream>>>(bQ, bK, bV, bS, bOffs, bCsrc, nullptr, bXH);

    // layer 2
    k_gemm<<<gemmBlocks, 256, 0, stream>>>(bXH, bW[4], bl[4], bQ, NNODES, DIM, DIM);
    k_gemm<<<gemmBlocks, 256, 0, stream>>>(bXH, bW[5], bl[5], bK, NNODES, DIM, DIM);
    k_gemm<<<gemmBlocks, 256, 0, stream>>>(bXH, bW[6], bl[6], bV, NNODES, DIM, DIM);
    k_gemm<<<gemmBlocks, 256, 0, stream>>>(bXH, bW[7], bl[7], bS, NNODES, DIM, DIM);
    k_edge_attn<<<NNODES, 256, 0, stream>>>(bQ, bK, bV, bS, bOffs, bCsrc, bH2, bXH);

    // attention pooling
    k_gather_q<<<(NB * DIM + 255) / 256, 256, 0, stream>>>(bXH, bQh);
    k_gemm<<<gemmBlocksQ, 256, 0, stream>>>(bQh, bW[8], nullptr, bQpart, NB, DIM, DIM);
    k_gemm<<<gemmBlocks, 256, 0, stream>>>(bXH, bW[9], atti_b, bQ /* p_part, reuse */, NNODES, DIM, DIM);
    k_score<<<NNODES / 8, 256, 0, stream>>>(bQ, bQpart, atts_w, atts_b, bScore);
    k_pool<<<NB, 256, 0, stream>>>(bScore, bH2, bPool);
    k_head<<<NB, 256, 0, stream>>>(bPool, fc1_w, fc1_b, fc2_w, fc2_b, (float*)d_out);
}
